// TimeFrequencyScatteringLoss_85349590106582
// MI455X (gfx1250) — matrix operand order repaired and checked by reading, not compiled, not run on hardware
//
#include <hip/hip_runtime.h>
#include <stdint.h>

// ---------------------------------------------------------------------------
// Time-frequency scattering loss on MI455X (gfx1250, wave32).
// 16K FFT = four-step (128x128): two complex 128^3 matmuls on the WMMA pipe
// (V_WMMA_F32_16X16X32_BF16, split-bf16 hi+lo fp32 emulation) + twiddles.
// LDS data: four bf16 planes (reHi/reLo/imHi/imLo, padded stride) typed as
// explicit address_space(3) pointers -> ds ops with 32-bit address math, no
// generic-pointer null guards in the WMMA loop.  Only scalar element types
// live in LDS (HIP vector classes cannot bind AS(3) references).
// ---------------------------------------------------------------------------

#define DIM   128
#define NTOT  16384
#define ROWP  132                      // padded LDS row stride (bank-safe)
#define PL    (DIM * ROWP)             // plane length in elements (16896)
#define PI2   6.283185307179586f
#define AS3   __attribute__((address_space(3)))

typedef __attribute__((ext_vector_type(16))) __bf16 v16bf;
typedef __attribute__((ext_vector_type(16), aligned(4))) __bf16 v16bfu;
typedef __attribute__((ext_vector_type(8)))  float  v8f;

// ---------------- bf16 helpers (bit-level, RNE) -----------------------------
__device__ __forceinline__ unsigned short f2bf(float x) {
  unsigned u = __builtin_bit_cast(unsigned, x);
  u += 0x7fffu + ((u >> 16) & 1u);
  return (unsigned short)(u >> 16);
}
__device__ __forceinline__ float bf2f(unsigned short b) {
  return __builtin_bit_cast(float, ((unsigned)b) << 16);
}
__device__ __forceinline__ __bf16 mkbf(unsigned v) {
  return __builtin_bit_cast(__bf16, (unsigned short)(v & 0xffffu));
}
__device__ __forceinline__ float bfe(__bf16 b) {
  return bf2f(__builtin_bit_cast(unsigned short, b));
}

// ---------------- buffers: four bf16 planes in LDS --------------------------
struct CBuf { AS3 __bf16 *rh, *rl, *ih, *il; };

__device__ __forceinline__ float rdre(const CBuf& b, int i) { return bfe(b.rh[i]) + bfe(b.rl[i]); }
__device__ __forceinline__ float rdim(const CBuf& b, int i) { return bfe(b.ih[i]) + bfe(b.il[i]); }
__device__ __forceinline__ void store4(const CBuf& d, int idx, float re, float im) {
  unsigned short h = f2bf(re);
  d.rh[idx] = mkbf(h); d.rl[idx] = mkbf(f2bf(re - bf2f(h)));
  unsigned short g = f2bf(im);
  d.ih[idx] = mkbf(g); d.il[idx] = mkbf(f2bf(im - bf2f(g)));
}
// 16 consecutive bf16 (4-byte aligned) -> one WMMA operand vector, mov-free.
__device__ __forceinline__ v16bf load16(const AS3 __bf16* p) {
  return *(const AS3 v16bfu*)p;
}

// ---------------- WMMA ------------------------------------------------------
struct Frag { v16bf rh, rl, ih, il; };

__device__ __forceinline__ v8f wmma_acc(v16bf a, v16bf b, v8f c) {
  return __builtin_amdgcn_wmma_f32_16x16x32_bf16(false, a, false, b, (short)0, c,
                                                 false, false);
}
// 12 WMMAs per complex 16x16x32 step (split-bf16: hh + hl + lh per product).
__device__ __forceinline__ void cwmma(const Frag& a, const Frag& b,
                                      v8f& rr, v8f& ii, v8f& ci) {
  rr = wmma_acc(a.rh, b.rh, rr); rr = wmma_acc(a.rh, b.rl, rr); rr = wmma_acc(a.rl, b.rh, rr);
  ii = wmma_acc(a.ih, b.ih, ii); ii = wmma_acc(a.ih, b.il, ii); ii = wmma_acc(a.il, b.ih, ii);
  ci = wmma_acc(a.rh, b.ih, ci); ci = wmma_acc(a.rh, b.il, ci); ci = wmma_acc(a.rl, b.ih, ci);
  ci = wmma_acc(a.ih, b.rh, ci); ci = wmma_acc(a.ih, b.rl, ci); ci = wmma_acc(a.il, b.rh, ci);
}
__device__ __forceinline__ v8f v8zero() {
  v8f z;
#pragma unroll
  for (int i = 0; i < 8; ++i) z[i] = 0.0f;
  return z;
}

// ---------------- one 128x128 complex matmul stage: OUT = T * SRC -----------
// MODE: 0 = store natural, 1 = store transposed, 2 = |.|*scale stored natural
//       + magnitude sum, 3 = magnitude sum only (SRC of next use preserved).
// HASPSI: fold per-element filter (freq = col + 128*row) into the data load.
template <int MODE, bool HASPSI>
__device__ void stage_TxD(const CBuf src, const CBuf dst,
                          const AS3 unsigned* tabDc, const AS3 unsigned* tabDs,
                          unsigned conjmask, float scale, float* magsum,
                          const float* __restrict__ psi) {
  const int tid = threadIdx.x, lane = tid & 31, I = tid >> 5;   // 8 waves
  const int half = lane >> 4, m = lane & 15;

  // Hoisted DFT-table A fragments: depend only on (wave row-panel, kt).
  Frag af[4];
#pragma unroll
  for (int kt = 0; kt < 4; ++kt) {
    const int row = I * 16 + m;
#pragma unroll
    for (int i = 0; i < 16; ++i) {
      int k = half * 16 + i;
      int j = (row * (kt * 32 + k)) & (DIM - 1);
      unsigned dx = tabDc[j];
      unsigned dy = tabDs[j] ^ conjmask;
      af[kt].rh[i] = mkbf(dx); af[kt].rl[i] = mkbf(dx >> 16);
      af[kt].ih[i] = mkbf(dy); af[kt].il[i] = mkbf(dy >> 16);
    }
  }

  const int brow = (lane >> 4) << 4;  // B k base within 32-group (k = 16*half + i)
  float msum = 0.0f;
#pragma unroll 1
  for (int J = 0; J < 8; ++J) {
    v8f rr = v8zero(), ii = v8zero(), ci = v8zero();
#pragma unroll
    for (int kt = 0; kt < 4; ++kt) {
      const int krow = kt * 32 + brow;
      const int boff = krow * ROWP + J * 16 + m;
      Frag b;
      if (HASPSI) {                   // second-order: fold psi2 into the load
        const float* prow = psi + DIM * krow + J * 16 + m;
#pragma unroll
        for (int i = 0; i < 16; ++i) {
          float s  = prow[DIM * i];
          float re = rdre(src, boff + i * ROWP) * s;
          float im = rdim(src, boff + i * ROWP) * s;
          unsigned short h = f2bf(re), g = f2bf(im);
          b.rh[i] = mkbf(h); b.rl[i] = mkbf(f2bf(re - bf2f(h)));
          b.ih[i] = mkbf(g); b.il[i] = mkbf(f2bf(im - bf2f(g)));
        }
      } else {                        // strided ds loads down column n = m
#pragma unroll
        for (int i = 0; i < 16; ++i) {
          b.rh[i] = src.rh[boff + i * ROWP]; b.rl[i] = src.rl[boff + i * ROWP];
          b.ih[i] = src.ih[boff + i * ROWP]; b.il[i] = src.il[boff + i * ROWP];
        }
      }
      cwmma(af[kt], b, rr, ii, ci);
    }
    const int n = lane & 15;
#pragma unroll
    for (int r = 0; r < 8; ++r) {     // C row = 16I + r + 8*half, col = 16J + n
      float cr = rr[r] - ii[r];
      float cim = ci[r];
      int row = I * 16 + r + 8 * half, col = J * 16 + n;
      if (MODE == 0) {
        store4(dst, row * ROWP + col, cr, cim);
      } else if (MODE == 1) {
        store4(dst, col * ROWP + row, cr, cim);
      } else {
        float mag = __fsqrt_rn(cr * cr + cim * cim);
        msum += mag;
        if (MODE == 2) store4(dst, row * ROWP + col, mag * scale, 0.0f);
      }
    }
  }
  if (MODE >= 2) atomicAdd(magsum, msum);       // ds_add_f32 reduction
}

// pointwise twiddle W_N^(+-row*col), composed from two 128-entry (c,s) tables.
__device__ void twiddle_inplace(const CBuf buf, const AS3 float* t128,
                                const AS3 float* tN, float ssign) {
  for (int e = threadIdx.x; e < NTOT; e += blockDim.x) {
    int r = e >> 7, c = e & (DIM - 1);
    int idx = r * ROWP + c;
    int a = (r * c) & (NTOT - 1);
    int alo = a & (DIM - 1), ahi = a >> 7;
    float w1c = tN[2 * alo], w1s = tN[2 * alo + 1];
    float w2c = t128[2 * ahi], w2s = t128[2 * ahi + 1];
    float cw = w1c * w2c - w1s * w2s;
    float sw = (w1c * w2s + w1s * w2c) * ssign;
    float re = rdre(buf, idx), im = rdim(buf, idx);
    store4(buf, idx, re * cw - im * sw, re * sw + im * cw);
  }
}

__device__ void init_tables(AS3 float* t128, AS3 float* tN,
                            AS3 unsigned* tabDc, AS3 unsigned* tabDs) {
  for (int j = threadIdx.x; j < DIM; j += blockDim.x) {
    float s1, c1, sN, cN;
    __sincosf((PI2 / 128.0f) * (float)j, &s1, &c1);
    __sincosf((PI2 / 16384.0f) * (float)j, &sN, &cN);
    t128[2 * j] = c1; t128[2 * j + 1] = s1;
    tN[2 * j]   = cN; tN[2 * j + 1]   = sN;
    unsigned short ch = f2bf(c1), sh = f2bf(-s1);
    tabDc[j] = (unsigned)ch | ((unsigned)f2bf(c1 - bf2f(ch)) << 16);
    tabDs[j] = (unsigned)sh | ((unsigned)f2bf(-s1 - bf2f(sh)) << 16);
  }
}

// Forward: natural -> natural spectrum with freq(e) == e (stage1 stores
// transposed; symmetry of D keeps both stages in Table-x-Data form).
__device__ void fft_forward(const CBuf A, const CBuf B,
                            const AS3 unsigned* tabDc, const AS3 unsigned* tabDs,
                            const AS3 float* t128, const AS3 float* tN) {
  stage_TxD<1, false>(A, B, tabDc, tabDs, 0u, 0.0f, nullptr, nullptr);
  __syncthreads();
  twiddle_inplace(B, t128, tN, -1.0f);
  __syncthreads();
  stage_TxD<0, false>(B, A, tabDc, tabDs, 0u, 0.0f, nullptr, nullptr);
  __syncthreads();
}
// Inverse (unnormalized); optional psi on the input spectrum; MODE2 = 2 keeps
// |x|/N in A (u1), MODE2 = 3 only accumulates |x| (A's spectrum preserved).
template <int MODE2, bool HASPSI>
__device__ void fft_inverse(const CBuf A, const CBuf B,
                            const AS3 unsigned* tabDc, const AS3 unsigned* tabDs,
                            const AS3 float* t128, const AS3 float* tN,
                            const float* __restrict__ psi, float scale,
                            float* magsum) {
  stage_TxD<1, HASPSI>(A, B, tabDc, tabDs, 0x80008000u, 0.0f, nullptr, psi);
  __syncthreads();
  twiddle_inplace(B, t128, tN, +1.0f);
  __syncthreads();
  stage_TxD<MODE2, false>(B, A, tabDc, tabDs, 0x80008000u, scale, magsum, nullptr);
  __syncthreads();
}

// ---------------- LDS layout ------------------------------------------------
// planes (8*PL*2) | t128 (256 f32) | tN (256 f32) | tabDc (128) | tabDs (128)
#define TABOFF   (8 * PL * 2)
#define SMEM_BYTES (TABOFF + 2048 + 1024 + 16)        // ~273 KB < 320 KB WGP

#define CARVE_SMEM()                                                        \
  extern __shared__ char smem[];                                            \
  AS3 char* sb = (AS3 char*)smem;                                           \
  AS3 __bf16* pb = (AS3 __bf16*)sb;                                         \
  CBuf A = { pb, pb + PL, pb + 2 * PL, pb + 3 * PL };                       \
  CBuf Bb = { pb + 4 * PL, pb + 5 * PL, pb + 6 * PL, pb + 7 * PL };         \
  AS3 float* t128 = (AS3 float*)(sb + TABOFF);                              \
  AS3 float* tN   = t128 + 2 * DIM;                                         \
  AS3 unsigned* tabDc = (AS3 unsigned*)(tN + 2 * DIM);                      \
  AS3 unsigned* tabDs = tabDc + DIM;                                        \
  float* magsum = (float*)(smem + TABOFF + 2048 + 1024)

// ---------------- Tensor Data Mover staging (Xf block -> LDS) ---------------
#if defined(__has_builtin)
#if __has_builtin(__builtin_amdgcn_tensor_load_to_lds) && \
    __has_builtin(__builtin_amdgcn_s_wait_tensorcnt)
#define USE_TDM 1
#endif
#endif

#define XFBYTES (4 * PL * 2)          // 135168 B per signal (4 raw planes)

#ifdef USE_TDM
typedef unsigned int u32x4 __attribute__((ext_vector_type(4)));
typedef int          i32x4 __attribute__((ext_vector_type(4)));
typedef int          i32x8 __attribute__((ext_vector_type(8)));

__device__ __forceinline__ void tdm_load_1d(const void* gsrc, unsigned lds_off,
                                            unsigned nbytes) {
  unsigned long long ga = (unsigned long long)(uintptr_t)gsrc;
  unsigned n = nbytes >> 3;                        // 8-byte elements
  u32x4 g0;
  g0[0] = 1u;                                      // count=1, user descriptor
  g0[1] = lds_off;                                 // lds_addr
  g0[2] = (unsigned)ga;                            // global_addr[31:0]
  g0[3] = (unsigned)((ga >> 32) & 0x01ffffffu) | (2u << 30);  // hi bits, type=2
  i32x8 g1;
  g1[0] = (int)(3u << 16);                         // data_size = 8 bytes
  g1[1] = (int)((n & 0xffffu) << 16);              // tensor_dim0[15:0]
  g1[2] = (int)(((n >> 16) & 0xffffu) | (1u << 16)); // dim0 hi | tensor_dim1=1
  g1[3] = (int)((n & 0xffffu) << 16);              // tile_dim0 = n
  g1[4] = 0;
  g1[5] = (int)n;                                  // tensor_dim0_stride
  g1[6] = 0;
  g1[7] = 0;
  i32x4 z4; z4[0] = z4[1] = z4[2] = z4[3] = 0;
#if __clang_major__ <= 22
  __builtin_amdgcn_tensor_load_to_lds(g0, g1, z4, z4, 0);
#else
  i32x8 z8; z8[0]=z8[1]=z8[2]=z8[3]=z8[4]=z8[5]=z8[6]=z8[7]=0;
  __builtin_amdgcn_tensor_load_to_lds(g0, g1, z4, z4, z8, 0);
#endif
}
#endif

// ---------------- Kernels ---------------------------------------------------
__global__ __launch_bounds__(256) void k_synth(const float* __restrict__ yin,
                                               const float* __restrict__ smin,
                                               const float* __restrict__ smax,
                                               float* __restrict__ sig) {
  const int b = blockIdx.x;
  float th[5];
#pragma unroll
  for (int p = 0; p < 5; ++p)
    th[p] = (yin[b * 5 + p] + 1.0f) * 0.5f * (smax[p] - smin[p]) + smin[p];
  const float f0 = th[0], al = th[1], pw = th[2], am = th[3], be = th[4];
  for (int n = threadIdx.x; n < NTOT; n += blockDim.x) {
    float t = (float)n * (1.0f / (float)NTOT);
    float acc = 0.0f, w = 1.0f;
#pragma unroll
    for (int k = 1; k <= 8; ++k) {
      w *= am;
      float kf = (float)k;
      float env = __expf(-al * (1.0f + be * kf) * t);
      float car = __sinf(PI2 * f0 * __powf(kf, pw) * t);
      acc += w * env * car;
    }
    sig[b * NTOT + n] = acc;
  }
}

// forward FFT of each synth signal; raw 4-plane block written to global.
__global__ __launch_bounds__(256) void k_fft_fwd(const float* __restrict__ sig,
                                                 unsigned* __restrict__ Xf) {
  CARVE_SMEM();
  (void)magsum;
  init_tables(t128, tN, tabDc, tabDs);
  const int b = blockIdx.x;
  for (int e = threadIdx.x; e < NTOT; e += blockDim.x) {
    int idx = (e >> 7) * ROWP + (e & (DIM - 1));
    store4(A, idx, sig[b * NTOT + e], 0.0f);
  }
  __syncthreads();
  fft_forward(A, Bb, tabDc, tabDs, t128, tN);
  const AS3 unsigned* lw = (const AS3 unsigned*)sb;   // 4 planes, contiguous
  unsigned* gw = Xf + (size_t)b * (XFBYTES / 4);
  for (int w = threadIdx.x; w < XFBYTES / 4; w += blockDim.x) gw[w] = lw[w];
}

// one block per (b, j1): TDM-stage spectrum, ifft(X*psi1)->S1,u1,
// fft(u1)->U1f (stays in A), then 4x ifft(U1f*psi2) magnitudes -> S2.
__global__ __launch_bounds__(256) void k_scatter(const unsigned* __restrict__ Xf,
                                                 const float* __restrict__ psi1,
                                                 const float* __restrict__ psi2,
                                                 float* __restrict__ S) {
  CARVE_SMEM();
  const int b = blockIdx.x >> 5;
  const int j1 = blockIdx.x & 31;
  init_tables(t128, tN, tabDc, tabDs);
  if (threadIdx.x == 0) *magsum = 0.0f;

#ifdef USE_TDM
  if (threadIdx.x == 0)
    tdm_load_1d(Xf + (size_t)b * (XFBYTES / 4), 0u, XFBYTES);
  __builtin_amdgcn_s_wait_tensorcnt(0);
#else
  {
    AS3 unsigned* lw = (AS3 unsigned*)sb;
    const unsigned* gw = Xf + (size_t)b * (XFBYTES / 4);
    for (int w = threadIdx.x; w < XFBYTES / 4; w += blockDim.x) lw[w] = gw[w];
  }
#endif
  __syncthreads();

  // Y = X .* psi1[j1]; spectrum is in natural frequency order (freq == e).
  for (int e = threadIdx.x; e < NTOT; e += blockDim.x) {
    int idx = (e >> 7) * ROWP + (e & (DIM - 1));
    float s = psi1[j1 * NTOT + e];
    store4(A, idx, rdre(A, idx) * s, rdim(A, idx) * s);
  }
  __syncthreads();

  const float invN = 1.0f / (float)NTOT;
  fft_inverse<2, false>(A, Bb, tabDc, tabDs, t128, tN, nullptr, invN, magsum);
  if (threadIdx.x == 0) {
    S[b * 160 + j1] = (*magsum) * invN * invN;
    *magsum = 0.0f;
  }
  __syncthreads();

  fft_forward(A, Bb, tabDc, tabDs, t128, tN);   // A = U1f (preserved below)

  for (int j2 = 0; j2 < 4; ++j2) {
    fft_inverse<3, true>(A, Bb, tabDc, tabDs, t128, tN, psi2 + j2 * NTOT,
                         0.0f, magsum);
    if (threadIdx.x == 0) {
      S[b * 160 + 32 + j1 * 4 + j2] = (*magsum) * invN * invN;
      *magsum = 0.0f;
    }
    __syncthreads();
  }
}

// proj[b][p] = nansum_d J[b][p][d]*(Sx-Sy)[b][d];  out = ||proj||_F
__global__ __launch_bounds__(128) void k_loss(const float* __restrict__ Sx,
                                              const float* __restrict__ Sy,
                                              const float* __restrict__ Jm,
                                              float* __restrict__ out) {
  __shared__ float acc;
  if (threadIdx.x == 0) acc = 0.0f;
  __syncthreads();
  if (threadIdx.x < 80) {
    int b = threadIdx.x / 5, p = threadIdx.x % 5;
    float s = 0.0f;
    for (int d = 0; d < 160; ++d) {
      float t = Jm[(b * 5 + p) * 160 + d] * (Sx[b * 160 + d] - Sy[b * 160 + d]);
      if (t == t) s += t;                       // nansum semantics
    }
    atomicAdd(&acc, s * s);
  }
  __syncthreads();
  if (threadIdx.x == 0) out[0] = __fsqrt_rn(acc);
}

// ---------------- Host launch ----------------------------------------------
extern "C" void kernel_launch(void* const* d_in, const int* in_sizes, int n_in,
                              void* d_out, int out_size, void* d_ws, size_t ws_size,
                              hipStream_t stream) {
  (void)in_sizes; (void)n_in; (void)out_size; (void)ws_size;
  const float* x    = (const float*)d_in[0];
  const float* y    = (const float*)d_in[1];
  const float* Jm   = (const float*)d_in[2];
  const float* smin = (const float*)d_in[3];
  const float* smax = (const float*)d_in[4];
  const float* psi1 = (const float*)d_in[5];
  const float* psi2 = (const float*)d_in[6];

  char*     ws  = (char*)d_ws;
  float*    sig = (float*)ws;                       // 16*16384 f32      (1 MB)
  unsigned* Xf  = (unsigned*)(ws + (1u << 20));     // 16 * 135168 B  (2.07 MB)
  float*    Sx  = (float*)(ws + 4u * (1u << 20));   // 16*160 f32
  float*    Sy  = Sx + 16 * 160;

  (void)hipFuncSetAttribute((const void*)k_fft_fwd,
                            hipFuncAttributeMaxDynamicSharedMemorySize,
                            SMEM_BYTES);
  (void)hipFuncSetAttribute((const void*)k_scatter,
                            hipFuncAttributeMaxDynamicSharedMemorySize,
                            SMEM_BYTES);

  for (int c = 0; c < 2; ++c) {
    const float* in = (c == 0) ? x : y;
    float* S = (c == 0) ? Sx : Sy;
    k_synth<<<16, 256, 0, stream>>>(in, smin, smax, sig);
    k_fft_fwd<<<16, 256, SMEM_BYTES, stream>>>(sig, Xf);
    k_scatter<<<512, 256, SMEM_BYTES, stream>>>(Xf, psi1, psi2, S);
  }
  k_loss<<<1, 128, 0, stream>>>(Sx, Sy, Jm, (float*)d_out);
}
